// AdvancedFraudDetectionModel_70385924047305
// MI455X (gfx1250) — compile-verified, matrix operand order checked by reading, not run on hardware
//
#include <hip/hip_runtime.h>
#include <hip/hip_bf16.h>

// ---------------- model constants (from reference) ----------------
#define NN      40960      // B*T nodes
#define DIN     256
#define HID     128
#define NB      4096       // batch
#define TT_     10         // seq len
#define NE      655360     // edges
#define HEADS   2

typedef __attribute__((ext_vector_type(16))) _Float16 v16h;
typedef __attribute__((ext_vector_type(8)))  _Float16 v8h;
typedef __attribute__((ext_vector_type(8)))  float    v8f;

// =================================================================
//  WMMA GEMM:  C[M,N] = A[M,K] * B(K,N)  (+C if accumulate)
//  A row-major fp32 (row stride lda). BT=0: B[k*ldb+n]; BT=1: B is
//  (N,K) row-major -> element (k,n) = B[n*ldb+k]  (used for W^T).
//  Block = 128 threads = 4 waves; block computes a 64(M) x 16(N)
//  macro-tile (wave w owns rows 16w..16w+15).
//  BT=1: B tile staged fp32 via GLOBAL_LOAD_ASYNC_TO_LDS_B128
//        (ASYNCcnt + s_wait_asynccnt), consumed as ds_load_b128.
//  BT=0: B tile staged f16 synchronously (rows contiguous in N).
//  Requires: M%64==0, N%16==0, K%32==0 (true for all call sites).
// =================================================================
template <int BT>
__launch_bounds__(128)
__global__ void gemm_wmma(const float* __restrict__ A, int lda,
                          const float* __restrict__ B, int ldb,
                          float* __restrict__ C, int ldc,
                          int M, int N, int K, int accumulate)
{
    // shared tile: BT=1 -> float [16][36] (rows 144B); BT=0 -> f16 [16][40]
    __shared__ __align__(16) unsigned char smem[16 * 36 * 4];
    float    (*Btf)[36] = (float(*)[36])smem;
    _Float16 (*Bth)[40] = (_Float16(*)[40])smem;

    const int lane    = threadIdx.x & 31;
    const int wave    = threadIdx.x >> 5;
    const int tilesN  = N >> 4;
    const int tm64    = blockIdx.x / tilesN;   // 64-row macro tile
    const int tn      = blockIdx.x % tilesN;
    const int tnBase  = tn << 4;

    const int halfRow = lane & 15;             // 0..15
    const int koff    = (lane >> 4) << 3;      // 0 or 8 (K half select)
    const int arow    = tm64 * 64 + wave * 16 + halfRow;

    v8f acc = {};
    const int cN     = tnBase + halfRow;
    const int cMbase = tm64 * 64 + wave * 16 + ((lane >> 4) << 3);
    if (accumulate) {
#pragma unroll
        for (int r = 0; r < 8; ++r)
            acc[r] = C[(size_t)(cMbase + r) * ldc + cN];
    }

    // per-thread staging coordinates
    const int bn   = threadIdx.x >> 3;          // BT=1: row n  (0..15)
    const int bk   = (threadIdx.x & 7) << 2;    // BT=1: k chunk (0,4,..,28)
    const int skk  = threadIdx.x >> 2;          // BT=0: k row  (0..31)
    const int sn   = (threadIdx.x & 3) << 2;    // BT=0: n chunk (0,4,8,12)
    const unsigned ldsDst = BT ? (unsigned)(size_t)&Btf[bn][bk] : 0u;
    const float*   gBrow  = BT ? (B + (size_t)(tnBase + bn) * ldb + bk) : nullptr;

    for (int k0 = 0; k0 < K; k0 += 32) {
        __syncthreads();                       // previous-iteration reads done
        if (BT) {
            // ---- async DMA: 16B per thread, global -> LDS, fp32 tile ----
            unsigned long long ga = (unsigned long long)(size_t)(gBrow + k0);
            asm volatile("global_load_async_to_lds_b128 %0, %1, off"
                         :: "v"(ldsDst), "v"(ga) : "memory");
            asm volatile("s_wait_asynccnt 0x0" ::: "memory");
        } else {
            // ---- sync stage, convert to f16: B[k*ldb+n], contiguous in n ----
            float4 v = *(const float4*)(B + (size_t)(k0 + skk) * ldb + tnBase + sn);
            Bth[sn + 0][skk] = (_Float16)v.x; Bth[sn + 1][skk] = (_Float16)v.y;
            Bth[sn + 2][skk] = (_Float16)v.z; Bth[sn + 3][skk] = (_Float16)v.w;
        }
        __syncthreads();

        // ---- A fragment: two contiguous 8-float spans per lane ----
        const float* arp = A + (size_t)arow * lda + k0;
        float4 a0 = *(const float4*)(arp + koff);
        float4 a1 = *(const float4*)(arp + koff + 4);
        float4 a2 = *(const float4*)(arp + 16 + koff);
        float4 a3 = *(const float4*)(arp + 16 + koff + 4);
        if (k0 + 32 < K)
            __builtin_prefetch(arp + 32, 0, 3);
        v16h av;
        av[0]  = (_Float16)a0.x; av[1]  = (_Float16)a0.y;
        av[2]  = (_Float16)a0.z; av[3]  = (_Float16)a0.w;
        av[4]  = (_Float16)a1.x; av[5]  = (_Float16)a1.y;
        av[6]  = (_Float16)a1.z; av[7]  = (_Float16)a1.w;
        av[8]  = (_Float16)a2.x; av[9]  = (_Float16)a2.y;
        av[10] = (_Float16)a2.z; av[11] = (_Float16)a2.w;
        av[12] = (_Float16)a3.x; av[13] = (_Float16)a3.y;
        av[14] = (_Float16)a3.z; av[15] = (_Float16)a3.w;

        // ---- B fragment from LDS ----
        v16h bv;
        if (BT) {
            float4 b0 = *(const float4*)(&Btf[halfRow][koff]);
            float4 b1 = *(const float4*)(&Btf[halfRow][koff + 4]);
            float4 b2 = *(const float4*)(&Btf[halfRow][16 + koff]);
            float4 b3 = *(const float4*)(&Btf[halfRow][16 + koff + 4]);
            bv[0]  = (_Float16)b0.x; bv[1]  = (_Float16)b0.y;
            bv[2]  = (_Float16)b0.z; bv[3]  = (_Float16)b0.w;
            bv[4]  = (_Float16)b1.x; bv[5]  = (_Float16)b1.y;
            bv[6]  = (_Float16)b1.z; bv[7]  = (_Float16)b1.w;
            bv[8]  = (_Float16)b2.x; bv[9]  = (_Float16)b2.y;
            bv[10] = (_Float16)b2.z; bv[11] = (_Float16)b2.w;
            bv[12] = (_Float16)b3.x; bv[13] = (_Float16)b3.y;
            bv[14] = (_Float16)b3.z; bv[15] = (_Float16)b3.w;
        } else {
            v8h blo = *(const v8h*)(&Bth[halfRow][koff]);
            v8h bhi = *(const v8h*)(&Bth[halfRow][16 + koff]);
#pragma unroll
            for (int i = 0; i < 8; ++i) { bv[i] = blo[i]; bv[8 + i] = bhi[i]; }
        }

        acc = __builtin_amdgcn_wmma_f32_16x16x32_f16(
                  false, av, false, bv, (short)0, acc, false, false);
    }
#pragma unroll
    for (int r = 0; r < 8; ++r)
        C[(size_t)(cMbase + r) * ldc + cN] = acc[r];
}

// ---------------- small elementwise helpers ----------------
__global__ void fill_f32(float* p, long long n, float v) {
    long long i = (long long)blockIdx.x * blockDim.x + threadIdx.x;
    if (i < n) p[i] = v;
}
__global__ void copy_f32(const float* __restrict__ s, float* __restrict__ d, long long n) {
    long long i = (long long)blockIdx.x * blockDim.x + threadIdx.x;
    if (i < n) d[i] = s[i];
}
// act: 0=none 1=relu 2=tanh
__global__ void bias_act(const float* __restrict__ in, float* __restrict__ out,
                         const float* __restrict__ bias, long long rows, int cols, int act) {
    long long i = (long long)blockIdx.x * blockDim.x + threadIdx.x;
    if (i >= rows * cols) return;
    int f = (int)(i % cols);
    float v = in[i] + bias[f];
    if (act == 1) v = fmaxf(v, 0.f);
    else if (act == 2) v = tanhf(v);
    out[i] = v;
}

// ---------------- GCN ----------------
__global__ void deg_edges(const int* __restrict__ dst, int ne, float* deg) {
    int e = blockIdx.x * blockDim.x + threadIdx.x;
    if (e < ne) atomicAdd(&deg[dst[e]], 1.0f);
}
__global__ void rsqrt_inplace(float* p, int n) {
    int i = blockIdx.x * blockDim.x + threadIdx.x;
    if (i < n) p[i] = rsqrtf(p[i]);
}
// thread per (edge, 4-float chunk); HID/4 = 32 chunks
__global__ void gcn_agg(const float* __restrict__ xw,
                        const int* __restrict__ src, const int* __restrict__ dst,
                        const float* __restrict__ inv, float* __restrict__ out,
                        long long total) {
    long long i = (long long)blockIdx.x * blockDim.x + threadIdx.x;
    if (i >= total) return;
    int c = (int)(i & 31);
    long long e = i >> 5;
    int s = src[e], d = dst[e];
    float nrm = inv[s] * inv[d];
    float4 v = ((const float4*)(xw + (size_t)s * HID))[c];
    float* o = out + (size_t)d * HID + c * 4;
    atomicAdd(o + 0, v.x * nrm); atomicAdd(o + 1, v.y * nrm);
    atomicAdd(o + 2, v.z * nrm); atomicAdd(o + 3, v.w * nrm);
}
__global__ void gcn_fin(const float* __restrict__ accb, const float* __restrict__ xw,
                        const float* __restrict__ inv, const float* __restrict__ bias,
                        float* __restrict__ out) {
    long long i = (long long)blockIdx.x * blockDim.x + threadIdx.x;
    if (i >= (long long)NN * HID) return;
    int node = (int)(i >> 7), f = (int)(i & 127);
    float selfn = inv[node] * inv[node];
    float v = accb[i] + xw[i] * selfn + bias[f];
    out[i] = fmaxf(v, 0.f);
}

// ---------------- GAT ----------------
__device__ __forceinline__ unsigned f2ord(float f) {
    unsigned u = __float_as_uint(f);
    return (u & 0x80000000u) ? ~u : (u | 0x80000000u);
}
__device__ __forceinline__ float ord2f(unsigned u) {
    return (u & 0x80000000u) ? __uint_as_float(u & 0x7FFFFFFFu) : __uint_as_float(~u);
}
__global__ void gat_al(const float* __restrict__ hh, const float* __restrict__ a_src,
                       const float* __restrict__ a_dst, float* als, float* ald) {
    int i = blockIdx.x * blockDim.x + threadIdx.x;
    if (i >= NN * HEADS) return;
    int node = i >> 1, hd = i & 1;
    const float* hv = hh + (size_t)node * (HEADS * HID) + hd * HID;
    const float* as = a_src + hd * HID;
    const float* ad = a_dst + hd * HID;
    float s = 0.f, d = 0.f;
    for (int k = 0; k < HID; ++k) { float v = hv[k]; s += v * as[k]; d += v * ad[k]; }
    als[i] = s; ald[i] = d;
}
// e >= NE encodes self loop for node (e-NE)
__device__ __forceinline__ float gat_eval(const int* src, const int* dst,
                                          const float* als, const float* ald,
                                          long long e, int hd, int& so, int& dd) {
    int s, d;
    if (e < NE) { s = src[e]; d = dst[e]; } else { s = d = (int)(e - NE); }
    so = s; dd = d;
    float v = als[s * 2 + hd] + ald[d * 2 + hd];
    return v > 0.f ? v : 0.2f * v;
}
__global__ void gat_max(const int* src, const int* dst, const float* als,
                        const float* ald, unsigned* m, long long total) {
    long long i = (long long)blockIdx.x * blockDim.x + threadIdx.x;
    if (i >= total) return;
    int hd = (int)(i & 1); long long e = i >> 1; int s, d;
    float v = gat_eval(src, dst, als, ald, e, hd, s, d);
    atomicMax(&m[d * 2 + hd], f2ord(v));
}
__global__ void gat_den(const int* src, const int* dst, const float* als,
                        const float* ald, const unsigned* m, float* den, long long total) {
    long long i = (long long)blockIdx.x * blockDim.x + threadIdx.x;
    if (i >= total) return;
    int hd = (int)(i & 1); long long e = i >> 1; int s, d;
    float v = gat_eval(src, dst, als, ald, e, hd, s, d);
    atomicAdd(&den[d * 2 + hd], expf(v - ord2f(m[d * 2 + hd])));
}
__global__ void gat_acc(const float* __restrict__ hh, const int* src, const int* dst,
                        const float* als, const float* ald, const unsigned* m,
                        const float* den, float* __restrict__ out, long long total) {
    long long i = (long long)blockIdx.x * blockDim.x + threadIdx.x;
    if (i >= total) return;
    int c = (int)(i & 31);
    long long r = i >> 5;
    int hd = (int)(r & 1); long long e = r >> 1; int s, d;
    float v = gat_eval(src, dst, als, ald, e, hd, s, d);
    float w = expf(v - ord2f(m[d * 2 + hd])) / den[d * 2 + hd] * 0.5f; // mean over 2 heads
    float4 hv = ((const float4*)(hh + (size_t)s * (HEADS * HID) + hd * HID))[c];
    float* o = out + (size_t)d * HID + c * 4;
    atomicAdd(o + 0, hv.x * w); atomicAdd(o + 1, hv.y * w);
    atomicAdd(o + 2, hv.z * w); atomicAdd(o + 3, hv.w * w);
}

// ---------------- LSTM cell ----------------
__device__ __forceinline__ float sigm(float x) { return 1.0f / (1.0f + expf(-x)); }
__global__ void lstm_cell(const float* __restrict__ G, const float* __restrict__ bih,
                          const float* __restrict__ bhh, float* __restrict__ hp,
                          float* __restrict__ cp, float* __restrict__ outb,
                          int tt, int dir_off) {
    int i = blockIdx.x * blockDim.x + threadIdx.x;
    if (i >= NB * HID) return;
    int b = i >> 7, j = i & 127;
    const float* g = G + (size_t)b * 512;
    float gi = g[j]       + bih[j]       + bhh[j];
    float gf = g[128 + j] + bih[128 + j] + bhh[128 + j];
    float gg = g[256 + j] + bih[256 + j] + bhh[256 + j];
    float go = g[384 + j] + bih[384 + j] + bhh[384 + j];
    float c  = sigm(gf) * cp[i] + sigm(gi) * tanhf(gg);
    float h  = sigm(go) * tanhf(c);
    cp[i] = c; hp[i] = h;
    outb[((size_t)b * TT_ + tt) * (2 * HID) + dir_off + j] = h;
}

// ---------------- attention / heads ----------------
__global__ void rowdot(const float* __restrict__ X, const float* __restrict__ w,
                       const float* __restrict__ b, float* out, int rows, int len) {
    int r = blockIdx.x * blockDim.x + threadIdx.x;
    if (r >= rows) return;
    float s = 0.f;
    const float* x = X + (size_t)r * len;
    for (int k = 0; k < len; ++k) s += x[k] * w[k];
    out[r] = s + b[0];
}
__global__ void attn_pool(const float* __restrict__ seq, const float* __restrict__ sc,
                          float* __restrict__ pooled) {
    int b = blockIdx.x, f = threadIdx.x;               // 4096 blocks x 256 threads
    const float* s = sc + b * TT_;
    float mx = s[0];
    for (int t = 1; t < TT_; ++t) mx = fmaxf(mx, s[t]);
    float w[TT_]; float sum = 0.f;
    for (int t = 0; t < TT_; ++t) { w[t] = expf(s[t] - mx); sum += w[t]; }
    float acc = 0.f;
    for (int t = 0; t < TT_; ++t)
        acc += (w[t] / sum) * seq[((size_t)b * TT_ + t) * (2 * HID) + f];
    pooled[(size_t)b * (2 * HID) + f] = acc;
}
// out[b*ncls+c] = dot(H[b,:128], W2[:,c]) + b2[c]
__global__ void head_out(const float* __restrict__ H, const float* __restrict__ W2,
                         const float* __restrict__ b2, float* out, int rows, int ncls) {
    int i = blockIdx.x * blockDim.x + threadIdx.x;
    if (i >= rows * ncls) return;
    int b = i / ncls, c = i % ncls;
    float s = 0.f;
    const float* h = H + (size_t)b * HID;
    for (int k = 0; k < HID; ++k) s += h[k] * W2[k * ncls + c];
    out[i] = s + b2[c];
}

// ---------------- host helpers ----------------
static inline int blks(long long n, int bs) { return (int)((n + bs - 1) / bs); }

static void gemm(const float* A, int lda, const float* B, int ldb, bool bT,
                 float* C, int ldc, int M, int N, int K, bool acc, hipStream_t st) {
    int grid = (M >> 6) * (N >> 4);            // 64x16 macro tiles
    if (bT)
        gemm_wmma<1><<<grid, 128, 0, st>>>(A, lda, B, ldb, C, ldc, M, N, K, acc ? 1 : 0);
    else
        gemm_wmma<0><<<grid, 128, 0, st>>>(A, lda, B, ldb, C, ldc, M, N, K, acc ? 1 : 0);
}

extern "C" void kernel_launch(void* const* d_in, const int* in_sizes, int n_in,
                              void* d_out, int out_size, void* d_ws, size_t ws_size,
                              hipStream_t stream) {
    (void)in_sizes; (void)n_in; (void)out_size; (void)ws_size;
    // ---- inputs (pytree-leaf order of setup_inputs) ----
    const float* x        = (const float*)d_in[0];
    const int*   ei       = (const int*)  d_in[1];            // (2,E)
    const int*   src      = ei;
    const int*   dst      = ei + NE;
    const float* gcn1W    = (const float*)d_in[4];
    const float* gcn1b    = (const float*)d_in[5];
    const float* gcn2W    = (const float*)d_in[6];
    const float* gcn2b    = (const float*)d_in[7];
    const float* gatW     = (const float*)d_in[8];
    const float* gatAs    = (const float*)d_in[9];
    const float* gatAd    = (const float*)d_in[10];
    const float* gatB     = (const float*)d_in[11];
    // lstm: fwd1 12-15, bwd1 16-19, fwd2 20-23, bwd2 24-27 (w_ih,w_hh,b_ih,b_hh)
    const float* attnW1   = (const float*)d_in[28];
    const float* attnB1   = (const float*)d_in[29];
    const float* attnW2   = (const float*)d_in[30];
    const float* attnB2   = (const float*)d_in[31];
    const float* embW     = (const float*)d_in[32];
    const float* embB     = (const float*)d_in[33];
    const float* clfW1    = (const float*)d_in[34];
    const float* clfB1    = (const float*)d_in[35];
    const float* clfW2    = (const float*)d_in[36];
    const float* clfB2    = (const float*)d_in[37];
    const float* anoW1    = (const float*)d_in[38];
    const float* anoB1    = (const float*)d_in[39];
    const float* anoW2    = (const float*)d_in[40];
    const float* anoB2    = (const float*)d_in[41];

    float* out = (float*)d_out;   // [0,8192) logits | [8192,12288) anomaly | [12288,..) emb

    // ---- workspace layout (floats) ----
    float* W = (float*)d_ws;
    size_t o = 0;
    auto alloc = [&](size_t n) { float* p = W + o; o += n; return p; };
    float* bigA   = alloc((size_t)NN * 256);       // xW / hh / tanh-proj
    float* hfeat  = alloc((size_t)NN * HID);       // node features
    float* accb   = alloc((size_t)NN * HID);       // scatter accumulator
    float* L1     = alloc((size_t)NB * TT_ * 256); // lstm layer1 out
    float* L2     = alloc((size_t)NB * TT_ * 256); // lstm layer2 out
    float* deg    = alloc(NN);
    float* als    = alloc((size_t)NN * HEADS);
    float* ald    = alloc((size_t)NN * HEADS);
    float* mbuf   = alloc((size_t)NN * HEADS);     // ordered-uint max
    float* den    = alloc((size_t)NN * HEADS);
    float* G      = alloc((size_t)NB * 512);
    float* hp     = alloc((size_t)NB * HID);
    float* cp     = alloc((size_t)NB * HID);
    float* scores = alloc((size_t)NN);
    float* pooled = alloc((size_t)NB * 256);
    float* emb    = alloc((size_t)NB * 64);
    float* hclf   = alloc((size_t)NB * HID);
    float* hano   = alloc((size_t)NB * HID);

    const long long NH  = (long long)NN * HID;
    const long long EA  = (long long)NE * 32;              // gcn scatter threads
    const long long GT  = ((long long)NE + NN) * HEADS;    // gat edges incl self
    const long long GTA = GT * 32;

    // ---- degree normalization (shared by both GCN layers) ----
    fill_f32<<<blks(NN, 256), 256, 0, stream>>>(deg, NN, 1.0f);   // self loop
    deg_edges<<<blks(NE, 256), 256, 0, stream>>>(dst, NE, deg);
    rsqrt_inplace<<<blks(NN, 256), 256, 0, stream>>>(deg, NN);    // deg -> 1/sqrt(deg)

    // ---- GCN layer 1 ----
    gemm(x, DIN, gcn1W, HID, false, bigA, HID, NN, HID, DIN, false, stream);
    fill_f32<<<blks(NH, 256), 256, 0, stream>>>(accb, NH, 0.f);
    gcn_agg<<<blks(EA, 256), 256, 0, stream>>>(bigA, src, dst, deg, accb, EA);
    gcn_fin<<<blks(NH, 256), 256, 0, stream>>>(accb, bigA, deg, gcn1b, hfeat);

    // ---- GCN layer 2 ----
    gemm(hfeat, HID, gcn2W, HID, false, bigA, HID, NN, HID, HID, false, stream);
    fill_f32<<<blks(NH, 256), 256, 0, stream>>>(accb, NH, 0.f);
    gcn_agg<<<blks(EA, 256), 256, 0, stream>>>(bigA, src, dst, deg, accb, EA);
    gcn_fin<<<blks(NH, 256), 256, 0, stream>>>(accb, bigA, deg, gcn2b, hfeat);

    // ---- GAT ----
    gemm(hfeat, HID, gatW, HEADS * HID, false, bigA, HEADS * HID, NN, HEADS * HID, HID, false, stream);
    gat_al<<<blks(NN * HEADS, 256), 256, 0, stream>>>(bigA, gatAs, gatAd, als, ald);
    fill_f32<<<blks(NN * HEADS, 256), 256, 0, stream>>>(mbuf, NN * HEADS, 0.f); // ord(-max)
    gat_max<<<blks(GT, 256), 256, 0, stream>>>(src, dst, als, ald, (unsigned*)mbuf, GT);
    fill_f32<<<blks(NN * HEADS, 256), 256, 0, stream>>>(den, NN * HEADS, 0.f);
    gat_den<<<blks(GT, 256), 256, 0, stream>>>(src, dst, als, ald, (const unsigned*)mbuf, den, GT);
    fill_f32<<<blks(NH, 256), 256, 0, stream>>>(accb, NH, 0.f);
    gat_acc<<<blks(GTA, 256), 256, 0, stream>>>(bigA, src, dst, als, ald,
                                                (const unsigned*)mbuf, den, accb, GTA);
    bias_act<<<blks(NH, 256), 256, 0, stream>>>(accb, hfeat, gatB, NN, HID, 0);

    // ---- bidirectional LSTM, 2 layers ----
    const int lbase[2][2] = { {12, 16}, {20, 24} };
    const float* lin[2]  = { hfeat, L1 };   // layer inputs
    float*       lout[2] = { L1, L2 };
    const int    indim[2] = { HID, 2 * HID };
    for (int layer = 0; layer < 2; ++layer) {
        for (int dir = 0; dir < 2; ++dir) {
            const float* w_ih = (const float*)d_in[lbase[layer][dir] + 0];
            const float* w_hh = (const float*)d_in[lbase[layer][dir] + 1];
            const float* b_ih = (const float*)d_in[lbase[layer][dir] + 2];
            const float* b_hh = (const float*)d_in[lbase[layer][dir] + 3];
            fill_f32<<<blks((long long)NB * HID, 256), 256, 0, stream>>>(hp, (long long)NB * HID, 0.f);
            fill_f32<<<blks((long long)NB * HID, 256), 256, 0, stream>>>(cp, (long long)NB * HID, 0.f);
            for (int step = 0; step < TT_; ++step) {
                int tt = dir ? (TT_ - 1 - step) : step;
                const float* A = lin[layer] + (size_t)tt * indim[layer];
                gemm(A, TT_ * indim[layer], w_ih, indim[layer], true,
                     G, 512, NB, 512, indim[layer], false, stream);
                gemm(hp, HID, w_hh, HID, true, G, 512, NB, 512, HID, true, stream);
                lstm_cell<<<blks((long long)NB * HID, 256), 256, 0, stream>>>(
                    G, b_ih, b_hh, hp, cp, lout[layer], tt, dir * HID);
            }
        }
    }

    // ---- attention pooling over T ----
    gemm(L2, 2 * HID, attnW1, HID, false, bigA, HID, NN, HID, 2 * HID, false, stream);
    bias_act<<<blks(NH, 256), 256, 0, stream>>>(bigA, bigA, attnB1, NN, HID, 2); // tanh
    rowdot<<<blks(NN, 256), 256, 0, stream>>>(bigA, attnW2, attnB2, scores, NN, HID);
    attn_pool<<<NB, 256, 0, stream>>>(L2, scores, pooled);

    // ---- embedding ----
    gemm(pooled, 2 * HID, embW, 64, false, emb, 64, NB, 64, 2 * HID, false, stream);
    bias_act<<<blks((long long)NB * 64, 256), 256, 0, stream>>>(emb, emb, embB, NB, 64, 0);
    copy_f32<<<blks((long long)NB * 64, 256), 256, 0, stream>>>(emb, out + 8192 + 4096, (long long)NB * 64);

    // ---- classifier head ----
    gemm(emb, 64, clfW1, HID, false, hclf, HID, NB, HID, 64, false, stream);
    bias_act<<<blks((long long)NB * HID, 256), 256, 0, stream>>>(hclf, hclf, clfB1, NB, HID, 1);
    head_out<<<blks(NB * 2, 256), 256, 0, stream>>>(hclf, clfW2, clfB2, out, NB, 2);

    // ---- anomaly head ----
    gemm(emb, 64, anoW1, HID, false, hano, HID, NB, HID, 64, false, stream);
    bias_act<<<blks((long long)NB * HID, 256), 256, 0, stream>>>(hano, hano, anoB1, NB, HID, 1);
    head_out<<<blks(NB, 256), 256, 0, stream>>>(hano, anoW2, anoB2, out + 8192, NB, 1);
}
